// STFT_12463995093476
// MI455X (gfx1250) — matrix operand order repaired and checked by reading, not compiled, not run on hardware
//
#include <hip/hip_runtime.h>

typedef float  v8f   __attribute__((ext_vector_type(8)));
typedef __bf16 v8bf  __attribute__((ext_vector_type(8)));
typedef __bf16 v16bf __attribute__((ext_vector_type(16)));
typedef unsigned int v4u __attribute__((ext_vector_type(4)));
typedef int v8i __attribute__((ext_vector_type(8)));
typedef int v4i __attribute__((ext_vector_type(4)));

#define NFILT     1024
#define HOP       512
#define PAD       512
#define NCH       1026                 // 2*CUTOFF
#define NS        480000
#define BATCH     32
#define T_FRAMES  938                  // (NS + 2*PAD - NFILT)/HOP + 1
#define L_FULL    480768               // NFILT + HOP*(T_FRAMES-1)
#define LP        (NS + 2 * PAD)       // 481024 padded length
#define NROWS     (BATCH * T_FRAMES)   // 30016 frames

#define TILE_R 64
#define TILE_C 64
#define KC     64
#define KPAD   72                      // LDS row stride in bf16 (144B, 16B-aligned)

static __device__ inline void split_bf16(float v, __bf16& hi, __bf16& lo) {
    hi = (__bf16)v;
    lo = (__bf16)(v - (float)hi);
}

// TDM: load a 64(rows) x 64(bf16) tile of a [*, NFILT] bf16 tensor into LDS,
// padding each 128B row to 144B to match the KPAD=72 LDS layout.
static __device__ inline void tdm_load_tile_bf16(const void* gptr, void* lptr) {
    const unsigned long long ga = (unsigned long long)(size_t)gptr;
    const unsigned lds = (unsigned)(size_t)lptr;     // low 32b = LDS offset
    v4u g0;
    g0.x = 1u;                                       // count=1 (valid), user D#
    g0.y = lds;                                      // lds_addr
    g0.z = (unsigned)ga;                             // global_addr[31:0]
    g0.w = (unsigned)((ga >> 32) & 0x01ffffffu)      // global_addr[56:32]
         | (2u << 30);                               // type=2 ("image")
    v8i g1;
    g1[0] = (1 << 16)                                // data_size=1 -> 2 bytes
          | (1 << 20)                                // pad_enable
          | (4 << 22)                                // pad_interval: 32 DWORDs
          | (3 << 25);                               // pad_amount: 4 DWORDs
    g1[1] = (NFILT & 0xffff) << 16;                  // tensor_dim0[15:0]
    g1[2] = (NFILT & 0xffff) << 16;                  // tensor_dim1[15:0]
    g1[3] = KC << 16;                                // tile_dim0 = 64
    g1[4] = TILE_C;                                  // tile_dim1 = 64
    g1[5] = NFILT;                                   // tensor_dim0_stride
    g1[6] = 0;
    g1[7] = 0;
    const v4i z4 = {0, 0, 0, 0};                     // 2D: groups 2/3 unused
    const v8i z8 = {0, 0, 0, 0, 0, 0, 0, 0};
    __builtin_amdgcn_tensor_load_to_lds(g0, g1, z4, z4, z8, 0);
}

// ---------------- kernel 0: zero the overlap-add accumulator ----------------
__global__ void zero_y_kernel(float* __restrict__ y, int n) {
    int i = blockIdx.x * blockDim.x + threadIdx.x;
    if (i < n) y[i] = 0.0f;
}

// ------- kernel A: materialize reflect-padded signal as bf16 hi/lo ----------
__global__ void prep_x_kernel(const float* __restrict__ x,
                              __bf16* __restrict__ xh,
                              __bf16* __restrict__ xl, int n) {
    int i = blockIdx.x * blockDim.x + threadIdx.x;
    if (i >= n) return;
    const int b = i / LP;
    const int p = i - b * LP;
    int s = p - PAD;
    s = (s < 0) ? -s : s;
    s = (s >= NS) ? (2 * NS - 2 - s) : s;
    const float v = x[b * NS + s];
    __bf16 hi, lo;
    split_bf16(v, hi, lo);
    xh[i] = hi;
    xl[i] = lo;
}

// ---- kernel 1: Mt[n][k] = sum_c fb[c][k]*inv[c][n], stored bf16 hi/lo ------
__global__ __launch_bounds__(256) void compute_M_kernel(
    const float* __restrict__ fb, const float* __restrict__ inv,
    __bf16* __restrict__ mth, __bf16* __restrict__ mtl) {
    __shared__ float sF[16][17];
    __shared__ float sI[16][17];
    const int k = blockIdx.y * 16 + threadIdx.y;
    const int n = blockIdx.x * 16 + threadIdx.x;
    float acc = 0.0f;
    for (int c0 = 0; c0 < NCH; c0 += 16) {
        const int c = c0 + threadIdx.y;
        sF[threadIdx.y][threadIdx.x] =
            (c < NCH) ? fb[c * NFILT + blockIdx.y * 16 + threadIdx.x] : 0.0f;
        sI[threadIdx.y][threadIdx.x] = (c < NCH) ? inv[c * NFILT + n] : 0.0f;
        __syncthreads();
#pragma unroll
        for (int cc = 0; cc < 16; ++cc)
            acc = fmaf(sF[cc][threadIdx.y], sI[cc][threadIdx.x], acc);
        __syncthreads();
    }
    __bf16 hi, lo;
    split_bf16(acc, hi, lo);
    mth[n * NFILT + k] = hi;          // transposed: row n, col k
    mtl[n * NFILT + k] = lo;
}

// -- kernel 2: framed GEMM, split-bf16 WMMA + TDM B-staging + overlap-add ----
__global__ __launch_bounds__(256) void gemm_ola_kernel(
    const __bf16* __restrict__ xh, const __bf16* __restrict__ xl,
    const __bf16* __restrict__ mth, const __bf16* __restrict__ mtl,
    float* __restrict__ y) {
    __shared__ __bf16 AsH[TILE_R][KPAD];
    __shared__ __bf16 AsL[TILE_R][KPAD];
    __shared__ __bf16 BtH[2][TILE_C][KPAD];
    __shared__ __bf16 BtL[2][TILE_C][KPAD];

    const int tid  = threadIdx.x;
    const int wave = tid >> 5;
    const int lane = tid & 31;
    const int h    = lane >> 4;            // lane half
    const int ln16 = lane & 15;

    const int rowBase = blockIdx.y * TILE_R;
    const int colBase = blockIdx.x * TILE_C;

    const int srow  = wave >> 1;           // 4x4 subtile grid, 2 subtiles/wave
    const int scol0 = (wave & 1) * 2;

    // A-staging addresses depend only on tid: rows r0 and r0+32, 8-elem group g
    const int r0 = tid >> 3;               // 0..31
    const int g  = (tid & 7) * 8;          // 0..56
    const int f0 = rowBase + r0;
    const int b0_ = f0 / T_FRAMES;
    const int t0_ = f0 - b0_ * T_FRAMES;
    const int f1 = f0 + 32;
    const int b1_ = f1 / T_FRAMES;
    const int t1_ = f1 - b1_ * T_FRAMES;
    const size_t offA0 = (size_t)b0_ * LP + t0_ * HOP + g;
    const size_t offA1 = (size_t)b1_ * LP + t1_ * HOP + g;

    const __bf16* mthTile = mth + (size_t)colBase * NFILT;
    const __bf16* mtlTile = mtl + (size_t)colBase * NFILT;

    v8f aHH0 = {}, aHL0 = {}, aLH0 = {};
    v8f aHH1 = {}, aHL1 = {}, aLH1 = {};

    // prologue: kick off TDM for chunk 0 into buffer 0
    if (wave == 0) {
        tdm_load_tile_bf16(mthTile, &BtH[0][0][0]);
        tdm_load_tile_bf16(mtlTile, &BtL[0][0][0]);
    }

    for (int c = 0; c < NFILT / KC; ++c) {
        const int k0 = c * KC;
        const int buf = c & 1;

        // stage A: straight-line 16B copies (no div, no loop)
        *(v8bf*)&AsH[r0][g]      = *(const v8bf*)(xh + offA0 + k0);
        *(v8bf*)&AsL[r0][g]      = *(const v8bf*)(xl + offA0 + k0);
        *(v8bf*)&AsH[r0 + 32][g] = *(const v8bf*)(xh + offA1 + k0);
        *(v8bf*)&AsL[r0 + 32][g] = *(const v8bf*)(xl + offA1 + k0);

        if (wave == 0) __builtin_amdgcn_s_wait_tensorcnt(0);
        __syncthreads();                       // A staged + B[buf] DMA done

        // overlap next B-panel DMA with this chunk's WMMAs
        if (wave == 0 && c + 1 < NFILT / KC) {
            tdm_load_tile_bf16(mthTile + (k0 + KC), &BtH[buf ^ 1][0][0]);
            tdm_load_tile_bf16(mtlTile + (k0 + KC), &BtL[buf ^ 1][0][0]);
        }

#pragma unroll
        for (int ks = 0; ks < KC; ks += 32) {
            // A frag 16x32: per lane 16 contiguous K starting at ks+h*16
            const int arow = srow * 16 + ln16;
            v8bf a0 = *(const v8bf*)&AsH[arow][ks + h * 16];
            v8bf a1 = *(const v8bf*)&AsH[arow][ks + h * 16 + 8];
            v16bf aH = __builtin_shufflevector(a0, a1, 0, 1, 2, 3, 4, 5, 6, 7,
                                               8, 9, 10, 11, 12, 13, 14, 15);
            a0 = *(const v8bf*)&AsL[arow][ks + h * 16];
            a1 = *(const v8bf*)&AsL[arow][ks + h * 16 + 8];
            v16bf aL = __builtin_shufflevector(a0, a1, 0, 1, 2, 3, 4, 5, 6, 7,
                                               8, 9, 10, 11, 12, 13, 14, 15);
            // B frag 32x16: per lane 16 contiguous K starting at ks+h*16
            const int bn0 = scol0 * 16 + ln16;
            v8bf b0 = *(const v8bf*)&BtH[buf][bn0][ks + h * 16];
            v8bf b1 = *(const v8bf*)&BtH[buf][bn0][ks + h * 16 + 8];
            v16bf bH0 = __builtin_shufflevector(b0, b1, 0, 1, 2, 3, 4, 5, 6, 7,
                                                8, 9, 10, 11, 12, 13, 14, 15);
            b0 = *(const v8bf*)&BtL[buf][bn0][ks + h * 16];
            b1 = *(const v8bf*)&BtL[buf][bn0][ks + h * 16 + 8];
            v16bf bL0 = __builtin_shufflevector(b0, b1, 0, 1, 2, 3, 4, 5, 6, 7,
                                                8, 9, 10, 11, 12, 13, 14, 15);
            b0 = *(const v8bf*)&BtH[buf][bn0 + 16][ks + h * 16];
            b1 = *(const v8bf*)&BtH[buf][bn0 + 16][ks + h * 16 + 8];
            v16bf bH1 = __builtin_shufflevector(b0, b1, 0, 1, 2, 3, 4, 5, 6, 7,
                                                8, 9, 10, 11, 12, 13, 14, 15);
            b0 = *(const v8bf*)&BtL[buf][bn0 + 16][ks + h * 16];
            b1 = *(const v8bf*)&BtL[buf][bn0 + 16][ks + h * 16 + 8];
            v16bf bL1 = __builtin_shufflevector(b0, b1, 0, 1, 2, 3, 4, 5, 6, 7,
                                                8, 9, 10, 11, 12, 13, 14, 15);

            aHH0 = __builtin_amdgcn_wmma_f32_16x16x32_bf16(
                false, aH, false, bH0, (short)0, aHH0, false, false);
            aHL0 = __builtin_amdgcn_wmma_f32_16x16x32_bf16(
                false, aH, false, bL0, (short)0, aHL0, false, false);
            aLH0 = __builtin_amdgcn_wmma_f32_16x16x32_bf16(
                false, aL, false, bH0, (short)0, aLH0, false, false);
            aHH1 = __builtin_amdgcn_wmma_f32_16x16x32_bf16(
                false, aH, false, bH1, (short)0, aHH1, false, false);
            aHL1 = __builtin_amdgcn_wmma_f32_16x16x32_bf16(
                false, aH, false, bL1, (short)0, aHL1, false, false);
            aLH1 = __builtin_amdgcn_wmma_f32_16x16x32_bf16(
                false, aL, false, bH1, (short)0, aLH1, false, false);
        }
        __syncthreads();                       // done reading As / Bt[buf]
    }

    const v8f s0 = aHH0 + aHL0 + aLH0;
    const v8f s1 = aHH1 + aHL1 + aLH1;

    // C/D layout: VGPR i, lanes0-15 -> row i; lanes16-31 -> row i+8
#pragma unroll
    for (int i = 0; i < 8; ++i) {
        const int r = srow * 16 + i + 8 * h;
        const int f = rowBase + r;
        const int b = f / T_FRAMES;
        const int t = f - b * T_FRAMES;
        float* ybase = y + (long)b * L_FULL + t * HOP;
        const int c0 = colBase + scol0 * 16 + ln16;
        atomicAdd(&ybase[c0],      s0[i]);
        atomicAdd(&ybase[c0 + 16], s1[i]);
    }
}

// ---------------- kernel 3: window-sum normalize, scale, crop ---------------
__global__ void finalize_kernel(const float* __restrict__ y,
                                const float* __restrict__ wsum,
                                float* __restrict__ out, int n) {
    int i = blockIdx.x * blockDim.x + threadIdx.x;
    if (i >= n) return;
    const int b = i / NS;
    const int s = i - b * NS;
    const int p = s + PAD;
    float v = y[(long)b * L_FULL + p];
    const float w = wsum[p];
    const float tiny = 1.1754943508222875e-38f;   // np.finfo(float32).tiny
    v = (w > tiny) ? (v / w) : v;
    out[i] = v * 2.0f;                            // * (NFILT / HOP)
}

extern "C" void kernel_launch(void* const* d_in, const int* in_sizes, int n_in,
                              void* d_out, int out_size, void* d_ws,
                              size_t ws_size, hipStream_t stream) {
    (void)in_sizes; (void)n_in; (void)out_size; (void)ws_size;
    const float* x    = (const float*)d_in[0];   // (32, 480000)
    const float* fb   = (const float*)d_in[1];   // (1026, 1, 1024)
    const float* inv  = (const float*)d_in[2];   // (1026, 1, 1024)
    const float* wsum = (const float*)d_in[3];   // (480768,)
    float* out = (float*)d_out;

    char* ws = (char*)d_ws;
    size_t o = 0;
    float*  y   = (float*)(ws + o);  o += (size_t)BATCH * L_FULL * 4;   // 61.5MB
    __bf16* mth = (__bf16*)(ws + o); o += (size_t)NFILT * NFILT * 2;    // 2MB
    __bf16* mtl = (__bf16*)(ws + o); o += (size_t)NFILT * NFILT * 2;    // 2MB
    __bf16* xh  = (__bf16*)(ws + o); o += (size_t)BATCH * LP * 2;       // 30.8MB
    __bf16* xl  = (__bf16*)(ws + o); o += (size_t)BATCH * LP * 2;       // 30.8MB

    const int yElems = BATCH * L_FULL;
    zero_y_kernel<<<(yElems + 255) / 256, 256, 0, stream>>>(y, yElems);

    const int xElems = BATCH * LP;
    prep_x_kernel<<<(xElems + 255) / 256, 256, 0, stream>>>(x, xh, xl, xElems);

    compute_M_kernel<<<dim3(NFILT / 16, NFILT / 16), dim3(16, 16), 0, stream>>>(
        fb, inv, mth, mtl);

    gemm_ola_kernel<<<dim3(NFILT / TILE_C, NROWS / TILE_R), 256, 0, stream>>>(
        xh, xl, mth, mtl, y);

    const int oElems = BATCH * NS;
    finalize_kernel<<<(oElems + 255) / 256, 256, 0, stream>>>(y, wsum, out,
                                                              oElems);
}
